// ProposalNetwork_70111046140296
// MI455X (gfx1250) — matrix operand order repaired and checked by reading, not compiled, not run on hardware
//
#include <hip/hip_runtime.h>

typedef __attribute__((ext_vector_type(16))) __bf16 v16bf;
typedef __attribute__((ext_vector_type(8)))  float  v8f;

#define BN_EPS 1e-3f

// ---------- helpers ----------
__device__ __forceinline__ unsigned pack_bf16x2(float a, float b) {
    unsigned ua = __float_as_uint(a);
    unsigned ub = __float_as_uint(b);
    ua += 0x7FFFu + ((ua >> 16) & 1u);   // round-to-nearest-even
    ub += 0x7FFFu + ((ub >> 16) & 1u);
    return (ua >> 16) | (ub & 0xFFFF0000u);
}
__device__ __forceinline__ unsigned short f32_to_bf16u(float f) {
    unsigned u = __float_as_uint(f);
    u += 0x7FFFu + ((u >> 16) & 1u);
    return (unsigned short)(u >> 16);
}
__device__ __forceinline__ float bf16u_to_f32(unsigned short h) {
    return __uint_as_float(((unsigned)h) << 16);
}

// ---------- elementwise f32 -> bf16 (2 elems/thread) ----------
__global__ __launch_bounds__(256)
void cvt_bf16_kernel(const float* __restrict__ X, unsigned* __restrict__ Y2, int n2) {
    int i = blockIdx.x * blockDim.x + threadIdx.x;
    if (i >= n2) return;
    float2 f = reinterpret_cast<const float2*>(X)[i];
    Y2[i] = pack_bf16x2(f.x, f.y);
}

// ---------- weight packing into WMMA B-fragment layout (bf16) ----------
// dword index = ((nt*8 + kt)*32 + lane)*8 + v
// lane L -> column n = nt*16 + (L&15); hi = L>>4
// dword v holds K = kt*32 + hi*8 + (v>>2)*8 + 2v (lo) and K+1 (hi)
__global__ __launch_bounds__(256)
void pack_w_kernel(const float* __restrict__ W, unsigned* __restrict__ Wp,
                   int ncols, int nTiles) {
    int idx = blockIdx.x * blockDim.x + threadIdx.x;
    int total = nTiles * 2048;
    if (idx >= total) return;
    int v    = idx & 7;
    int lane = (idx >> 3) & 31;
    int kt   = (idx >> 8) & 7;
    int nt   = idx >> 11;
    int n  = nt * 16 + (lane & 15);
    int K  = kt * 32 + (lane >> 4) * 8 + (v >> 2) * 8 + v * 2;
    float a0 = (n < ncols) ? W[(size_t)K       * ncols + n] : 0.0f;
    float a1 = (n < ncols) ? W[(size_t)(K + 1) * ncols + n] : 0.0f;
    Wp[idx] = pack_bf16x2(a0, a1);
}

// ---------- bf16 WMMA GEMM, K = 256, A row-major bf16, fused epilogue ----------
// MODE 0: BN -> bf16 out ; MODE 1: BN+ReLU -> bf16 out ; MODE 2: bias -> f32 out
// Each wave computes a 16 x (16*NTW) strip.
template<int NTW, int MODE>
__global__ __launch_bounds__(256)
void wmma_gemm_kernel(const unsigned short* __restrict__ Xb,
                      const unsigned* __restrict__ Wp,
                      unsigned short* __restrict__ Yb, float* __restrict__ Yf,
                      const float* __restrict__ p0, const float* __restrict__ p1,
                      const float* __restrict__ p2, const float* __restrict__ p3,
                      int mTiles, int nTiles, int ldY, int nCols) {
    int wave = threadIdx.x >> 5;
    int lane = threadIdx.x & 31;
    int strips = nTiles / NTW;
    int tile = blockIdx.x * (blockDim.x >> 5) + wave;
    if (tile >= mTiles * strips) return;
    int mt = tile / strips;
    int ns = tile - mt * strips;
    int l  = lane & 15;
    int hi = lane >> 4;

    // A row: contiguous bf16; fragment = 2 x uint4 at (kt*4+hi) and (kt*4+hi+2)
    const uint4* Arow = reinterpret_cast<const uint4*>(Xb + (size_t)(mt * 16 + l) * 256);
    const uint4* Wp4  = reinterpret_cast<const uint4*>(Wp);

    v8f acc[NTW];
    #pragma unroll
    for (int j = 0; j < NTW; ++j) acc[j] = v8f{};

    #pragma unroll
    for (int kt = 0; kt < 8; ++kt) {
        union { uint4 q[2]; v16bf v; } afr;
        afr.q[0] = Arow[kt * 4 + hi];
        afr.q[1] = Arow[kt * 4 + hi + 2];
        #pragma unroll
        for (int j = 0; j < NTW; ++j) {
            int nt = ns * NTW + j;
            union { uint4 q[2]; v16bf v; } bfr;
            size_t bidx = ((size_t)(nt * 8 + kt) * 32 + lane) * 2;
            bfr.q[0] = Wp4[bidx];
            bfr.q[1] = Wp4[bidx + 1];
            acc[j] = __builtin_amdgcn_wmma_f32_16x16x32_bf16(
                false, afr.v, false, bfr.v, (short)0, acc[j], false, false);
        }
    }

    // Epilogue. C layout: col = lane&15, row = r + 8*hi
    #pragma unroll
    for (int j = 0; j < NTW; ++j) {
        int col = (ns * NTW + j) * 16 + l;
        if (MODE == 2) {
            float bs = (col < nCols) ? p0[col] : 0.0f;
            #pragma unroll
            for (int r = 0; r < 8; ++r) {
                int m = mt * 16 + r + hi * 8;
                if (col < nCols) Yf[(size_t)m * ldY + col] = acc[j][r] + bs;
            }
        } else {
            float sc = p0[col] * rsqrtf(p3[col] + BN_EPS);
            float mn = p2[col];
            float bt = p1[col];
            #pragma unroll
            for (int r = 0; r < 8; ++r) {
                int m = mt * 16 + r + hi * 8;
                float y = (acc[j][r] - mn) * sc + bt;
                if (MODE == 1) y = fmaxf(y, 0.0f);
                Yb[(size_t)m * ldY + col] = f32_to_bf16u(y);
            }
        }
    }
}

// ---------- depthwise 5x5 cross-correlation (bf16 in, f32 + bf16 out) ----------
// feat (64,29,29,256) bf16, ker (64,5,5,256) bf16 -> corr f32 (d_out) + bf16 copy
__global__ __launch_bounds__(256)
void dwxcorr_kernel(const unsigned short* __restrict__ feat,
                    const unsigned short* __restrict__ ker,
                    float* __restrict__ corrF, unsigned short* __restrict__ corrB) {
    int b = blockIdx.x >> 2;
    int cchunk = blockIdx.x & 3;
    int clane = threadIdx.x & 63;
    int c = cchunk * 64 + clane;

    __shared__ float kl[25 * 64];
    for (int i = threadIdx.x; i < 25 * 64; i += 256) {
        int tap = i >> 6;
        int cc  = i & 63;
        kl[i] = bf16u_to_f32(ker[(size_t)(b * 25 + tap) * 256 + cchunk * 64 + cc]);
    }
    __syncthreads();

    const unsigned short* fb = feat + (size_t)b * 29 * 29 * 256 + c;
    size_t obase = (size_t)b * 25 * 25 * 256 + c;

    for (int p = (threadIdx.x >> 6); p < 625; p += 4) {
        int y = p / 25;
        int x = p - y * 25;
        float s = 0.0f;
        #pragma unroll
        for (int i = 0; i < 5; ++i) {
            #pragma unroll
            for (int j = 0; j < 5; ++j) {
                s = fmaf(bf16u_to_f32(fb[(size_t)((y + i) * 29 + (x + j)) * 256]),
                         kl[(i * 5 + j) * 64 + clane], s);
            }
        }
        corrF[obase + (size_t)p * 256] = s;
        corrB[obase + (size_t)p * 256] = f32_to_bf16u(s);
    }
}

// ---------- launch ----------
extern "C" void kernel_launch(void* const* d_in, const int* in_sizes, int n_in,
                              void* d_out, int out_size, void* d_ws, size_t ws_size,
                              hipStream_t stream) {
    const float* exemplar = (const float*)d_in[0];
    const float* search   = (const float*)d_in[1];
    const float* ex_w     = (const float*)d_in[2];
    const float* ex_g     = (const float*)d_in[3];
    const float* ex_b     = (const float*)d_in[4];
    const float* ex_m     = (const float*)d_in[5];
    const float* ex_v     = (const float*)d_in[6];
    const float* se_w     = (const float*)d_in[7];
    const float* se_g     = (const float*)d_in[8];
    const float* se_b     = (const float*)d_in[9];
    const float* se_m     = (const float*)d_in[10];
    const float* se_v     = (const float*)d_in[11];
    const float* c1_w     = (const float*)d_in[12];
    const float* c1_g     = (const float*)d_in[13];
    const float* c1_b     = (const float*)d_in[14];
    const float* c1_m     = (const float*)d_in[15];
    const float* c1_v     = (const float*)d_in[16];
    const float* c2_w     = (const float*)d_in[17];
    const float* c2_b     = (const float*)d_in[18];

    float* out   = (float*)d_out;                // (64,25,25,10)  = 400000
    float* corrF = out + 400000;                 // (64,25,25,256) = 10240000

    // workspace layout (float-unit offsets; all 16B aligned)
    float* ws = (float*)d_ws;
    unsigned* wp_ex = (unsigned*)(ws + 0);                 // 32768 u32
    unsigned* wp_se = (unsigned*)(ws + 32768);             // 32768 u32
    unsigned* wp_c1 = (unsigned*)(ws + 65536);             // 32768 u32
    unsigned* wp_c2 = (unsigned*)(ws + 98304);             //  2048 u32
    unsigned short* Xe    = (unsigned short*)(ws + 100352);   //   409600 u16
    unsigned short* Xs    = (unsigned short*)(ws + 202752);   // 13811712 u16
    unsigned short* kerB  = (unsigned short*)(ws + 3655680);  //   409600 u16
    unsigned short* featB = (unsigned short*)(ws + 3758080);  // 13811712 u16
    unsigned short* corrB = Xs;     // reuse: Xs dead after se GEMM (10240000 u16 fits)
    unsigned short* hB    = featB;  // reuse: feat dead after xcorr (10240000 u16 fits)

    const int M_EX = 64 * 25;    // 1600
    const int M_SE = 64 * 841;   // 53824
    const int M_CO = 64 * 625;   // 40000

    // 1) pack weights -> bf16 B fragments
    pack_w_kernel<<<128, 256, 0, stream>>>(ex_w, wp_ex, 256, 16);
    pack_w_kernel<<<128, 256, 0, stream>>>(se_w, wp_se, 256, 16);
    pack_w_kernel<<<128, 256, 0, stream>>>(c1_w, wp_c1, 256, 16);
    pack_w_kernel<<<  8, 256, 0, stream>>>(c2_w, wp_c2,  10,  1);

    // 2) activations -> bf16 row-major
    cvt_bf16_kernel<<<(M_EX * 128 + 255) / 256, 256, 0, stream>>>(exemplar, (unsigned*)Xe, M_EX * 128);
    cvt_bf16_kernel<<<(M_SE * 128 + 255) / 256, 256, 0, stream>>>(search,   (unsigned*)Xs, M_SE * 128);

    // 3) exemplar adjust: (1600x256)@(256x256) + BN -> kerB (bf16)
    wmma_gemm_kernel<4, 0><<<((M_EX / 16) * 4 + 7) / 8, 256, 0, stream>>>(
        Xe, wp_ex, kerB, nullptr, ex_g, ex_b, ex_m, ex_v, M_EX / 16, 16, 256, 256);

    // 4) search adjust: (53824x256)@(256x256) + BN -> featB (bf16)
    wmma_gemm_kernel<4, 0><<<((M_SE / 16) * 4 + 7) / 8, 256, 0, stream>>>(
        Xs, wp_se, featB, nullptr, se_g, se_b, se_m, se_v, M_SE / 16, 16, 256, 256);

    // 5) depthwise xcorr -> corr f32 (into d_out) + bf16 copy (reuses Xs space)
    dwxcorr_kernel<<<64 * 4, 256, 0, stream>>>(featB, kerB, corrF, corrB);

    // 6) c1: (40000x256)@(256x256) + BN + ReLU -> hB (bf16, reuses featB space)
    wmma_gemm_kernel<4, 1><<<((M_CO / 16) * 4 + 7) / 8, 256, 0, stream>>>(
        corrB, wp_c1, hB, nullptr, c1_g, c1_b, c1_m, c1_v, M_CO / 16, 16, 256, 256);

    // 7) c2: (40000x256)@(256x10) + bias -> out (f32)
    wmma_gemm_kernel<1, 2><<<((M_CO / 16) + 7) / 8, 256, 0, stream>>>(
        hB, wp_c2, nullptr, out, c2_b, c2_b, c2_b, c2_b, M_CO / 16, 1, 10, 10);
}
